// Net_90744069030470
// MI455X (gfx1250) — matrix operand order repaired and checked by reading, not compiled, not run on hardware
//
#include <hip/hip_runtime.h>
#include <hip/hip_bf16.h>
#include <math.h>

typedef __attribute__((ext_vector_type(16))) __bf16 v16bf;
typedef __attribute__((ext_vector_type(8)))  __bf16 v8bf;
typedef __attribute__((ext_vector_type(2)))  __bf16 v2bf;
typedef __attribute__((ext_vector_type(8)))  float  v8f;

#define WAVE 32

// ---------------------------------------------------------------------------
// fp32 -> bf16 conversion (2 elements/thread, packed store)
// ---------------------------------------------------------------------------
__global__ void cvt_f32_bf16(const float* __restrict__ in,
                             __bf16* __restrict__ out, long long n2)
{
    long long t = (long long)blockIdx.x * blockDim.x + threadIdx.x;
    if (t >= n2) return;
    float2 v = ((const float2*)in)[t];
    v2bf o; o[0] = (__bf16)v.x; o[1] = (__bf16)v.y;
    ((v2bf*)out)[t] = o;
}

// fp32 B[K,Ncols] -> bf16 Bt[Ncols,K] (transposed, coalesced writes)
__global__ void cvt_f32_bf16_T(const float* __restrict__ B,
                               __bf16* __restrict__ Bt, int K, int Ncols)
{
    int t = blockIdx.x * blockDim.x + threadIdx.x;
    if (t >= K * Ncols) return;
    int col = t / K, k = t % K;
    Bt[t] = (__bf16)B[(long long)k * Ncols + col];
}

// ---------------------------------------------------------------------------
// WMMA bf16 GEMM: C[M,Ncols] = A_bf16[M,K] * Bt_bf16[Ncols,K]^T (+bias)
// One wave computes a 16 x (16*NS) strip; A fragment reused across NS WMMAs.
// A frag per lane: halves K in [lhi*8, +8) and [16+lhi*8, +8)  (two b128 loads)
// B frag per lane: halves K in [lhi*8, +8) and [16+lhi*8, +8) of column (lane&15) (two b128 loads)
// ---------------------------------------------------------------------------
template<int NS>
__global__ void wmma_gemm_bf16(const __bf16* __restrict__ A,
                               const __bf16* __restrict__ Bt,
                               const float* __restrict__ bias,
                               float* __restrict__ C,
                               int Mtiles, int Ngroups, int K,
                               int lda, int ldc)
{
    int wave  = blockIdx.x * (blockDim.x / WAVE) + (threadIdx.x / WAVE);
    int total = Mtiles * Ngroups;
    if (wave >= total) return;                 // wave-uniform: EXEC stays all-1
    int tg   = wave % Ngroups;
    int tm   = wave / Ngroups;
    int lane = threadIdx.x & (WAVE - 1);
    int lhi  = lane >> 4;
    int llo  = lane & 15;

    const __bf16* arowp = A + (long long)(tm * 16 + llo) * lda;

    v8f acc[NS];
#pragma unroll
    for (int i = 0; i < NS; ++i) acc[i] = (v8f){};

    for (int k0 = 0; k0 < K; k0 += 32) {
        v8bf alo = *(const v8bf*)(arowp + k0 + lhi * 8);
        v8bf ahi = *(const v8bf*)(arowp + k0 + 16 + lhi * 8);
        v16bf av = __builtin_shufflevector(alo, ahi,
                     0,1,2,3,4,5,6,7,8,9,10,11,12,13,14,15);
#pragma unroll
        for (int i = 0; i < NS; ++i) {
            int bcol = (tg * NS + i) * 16 + llo;
            v8bf blo = *(const v8bf*)(Bt + (long long)bcol * K + k0 + lhi * 8);
            v8bf bhi = *(const v8bf*)(Bt + (long long)bcol * K + k0 + 16 + lhi * 8);
            v16bf bv = __builtin_shufflevector(blo, bhi,
                         0,1,2,3,4,5,6,7,8,9,10,11,12,13,14,15);
            acc[i] = __builtin_amdgcn_wmma_f32_16x16x32_bf16(false, av, false, bv,
                                                             (short)0, acc[i],
                                                             false, false);
        }
    }
#pragma unroll
    for (int i = 0; i < NS; ++i) {
        int col = (tg * NS + i) * 16 + llo;
        float bb = bias ? bias[col] : 0.f;     // hoisted: invariant over rows
#pragma unroll
        for (int r = 0; r < 8; ++r) {
            int row = tm * 16 + r + 8 * lhi;   // C/D: M = r + 8*(lane>>4)
            C[(long long)row * ldc + col] = acc[i][r] + bb;
        }
    }
}

// ---------------------------------------------------------------------------
// Framelet propagate, 256 features: out[dst] += ea[e] * x[src]  (float4 lanes)
// ---------------------------------------------------------------------------
__global__ void propagate256(const float* __restrict__ x,
                             const int* __restrict__ src,
                             const int* __restrict__ dst,
                             const float* __restrict__ ea,
                             float* __restrict__ out, int E)
{
    long long t = (long long)blockIdx.x * blockDim.x + threadIdx.x;
    if (t >= (long long)E * 64) return;
    int e  = (int)(t >> 6);
    int c4 = (int)(t & 63);
    int s = src[e], d = dst[e];
    float w = ea[e];
    float4 v = *((const float4*)(x + (long long)s * 256) + c4);
    float* o = out + (long long)d * 256 + c4 * 4;
    atomicAdd(o + 0, w * v.x);
    atomicAdd(o + 1, w * v.y);
    atomicAdd(o + 2, w * v.z);
    atomicAdd(o + 3, w * v.w);
}

// Framelet propagate, 16 features (accumulates y1+y2 directly into ysum)
__global__ void propagate16(const float* __restrict__ y,
                            const int* __restrict__ src,
                            const int* __restrict__ dst,
                            const float* __restrict__ ea,
                            float* __restrict__ out, int E)
{
    long long t = (long long)blockIdx.x * blockDim.x + threadIdx.x;
    if (t >= (long long)E * 16) return;
    int e = (int)(t >> 4);
    int c = (int)(t & 15);
    atomicAdd(&out[(long long)dst[e] * 16 + c],
              ea[e] * y[(long long)src[e] * 16 + c]);
}

// ---------------------------------------------------------------------------
// GAT attention scores: a_src/a_dst [N,H]
// ---------------------------------------------------------------------------
__global__ void gat_scores(const float* __restrict__ hg,
                           const float* __restrict__ att_src,
                           const float* __restrict__ att_dst,
                           float* __restrict__ asrc,
                           float* __restrict__ adst, int N)
{
    int t = blockIdx.x * blockDim.x + threadIdx.x;
    if (t >= N * 2) return;
    int n = t >> 1, hh = t & 1;
    const float* hp = hg + (long long)n * 32 + hh * 16;
    const float* as = att_src + hh * 16;
    const float* ad = att_dst + hh * 16;
    float s = 0.f, d = 0.f;
#pragma unroll
    for (int c = 0; c < 16; ++c) { float v = hp[c]; s += v * as[c]; d += v * ad[c]; }
    asrc[t] = s;
    adst[t] = d;
}

__device__ __forceinline__ void edgeSD(const int* eio, int E_O, int e, int& s, int& d)
{
    if (e < E_O) { s = eio[e]; d = eio[E_O + e]; }     // original edges
    else         { s = d = e - E_O; }                  // self loops
}

__device__ __forceinline__ void atomicMaxF(float* addr, float val)
{
    int* ai = (int*)addr;
    int cur = *ai;
    for (;;) {
        float f = __int_as_float(cur);
        if (f >= val) break;                            // NaN init -> replaced
        int prev = atomicCAS(ai, cur, __float_as_int(val));
        if (prev == cur) break;
        cur = prev;
    }
}

__global__ void gat_edge_max(const int* __restrict__ eio, int E_O, int Etot,
                             const float* __restrict__ asrc,
                             const float* __restrict__ adst,
                             float* __restrict__ mx)
{
    int t = blockIdx.x * blockDim.x + threadIdx.x;
    if (t >= Etot * 2) return;
    int e = t >> 1, hh = t & 1;
    int s, d; edgeSD(eio, E_O, e, s, d);
    float l = asrc[s * 2 + hh] + adst[d * 2 + hh];
    l = l > 0.f ? l : 0.2f * l;                         // leaky_relu
    atomicMaxF(&mx[d * 2 + hh], l);
}

__global__ void gat_edge_exp(const int* __restrict__ eio, int E_O, int Etot,
                             const float* __restrict__ asrc,
                             const float* __restrict__ adst,
                             const float* __restrict__ mx,
                             float* __restrict__ ebuf,
                             float* __restrict__ denom)
{
    int t = blockIdx.x * blockDim.x + threadIdx.x;
    if (t >= Etot * 2) return;
    int e = t >> 1, hh = t & 1;
    int s, d; edgeSD(eio, E_O, e, s, d);
    float l = asrc[s * 2 + hh] + adst[d * 2 + hh];
    l = l > 0.f ? l : 0.2f * l;
    float ex = __expf(l - mx[d * 2 + hh]);
    ebuf[t] = ex;
    atomicAdd(&denom[d * 2 + hh], ex);
}

__global__ void gat_aggregate(const int* __restrict__ eio, int E_O, int Etot,
                              const float* __restrict__ hg,
                              const float* __restrict__ ebuf,
                              const float* __restrict__ denom,
                              float* __restrict__ gout)
{
    long long t = (long long)blockIdx.x * blockDim.x + threadIdx.x;
    if (t >= (long long)Etot * 32) return;
    int e  = (int)(t >> 5);
    int j  = (int)(t & 31);
    int hh = j >> 4;
    int s, d; edgeSD(eio, E_O, e, s, d);
    float alpha = ebuf[e * 2 + hh] / denom[d * 2 + hh];
    atomicAdd(&gout[(long long)d * 32 + j], alpha * hg[(long long)s * 32 + j]);
}

// ---------------------------------------------------------------------------
// y = ELU(g + gat_bias) @ Wm + bm     [N,32] -> [N,16]
// ---------------------------------------------------------------------------
__global__ void mlp_elu(const float* __restrict__ gout,
                        const float* __restrict__ gb,
                        const float* __restrict__ Wm,
                        const float* __restrict__ bm,
                        float* __restrict__ y, int N)
{
    int t = blockIdx.x * blockDim.x + threadIdx.x;
    if (t >= N * 16) return;
    int n = t >> 4, c = t & 15;
    const float* g = gout + (long long)n * 32;
    float acc = bm[c];
#pragma unroll
    for (int j = 0; j < 32; ++j) {
        float v = g[j] + gb[j];
        v = v > 0.f ? v : (__expf(v) - 1.f);            // ELU
        acc += v * Wm[j * 16 + c];
    }
    y[t] = acc;
}

// ---------------------------------------------------------------------------
// Final log_softmax over 16 classes
// ---------------------------------------------------------------------------
__global__ void logsoftmax16(const float* __restrict__ ysum,
                             float* __restrict__ out, int N)
{
    int n = blockIdx.x * blockDim.x + threadIdx.x;
    if (n >= N) return;
    float v[16];
    float m = -3.0e38f;
#pragma unroll
    for (int c = 0; c < 16; ++c) { v[c] = ysum[(long long)n * 16 + c]; m = fmaxf(m, v[c]); }
    float s = 0.f;
#pragma unroll
    for (int c = 0; c < 16; ++c) s += __expf(v[c] - m);
    float lse = m + __logf(s);
#pragma unroll
    for (int c = 0; c < 16; ++c) out[(long long)n * 16 + c] = v[c] - lse;
}

// ===========================================================================
extern "C" void kernel_launch(void* const* d_in, const int* in_sizes, int n_in,
                              void* d_out, int out_size, void* d_ws, size_t ws_size,
                              hipStream_t stream)
{
    const float* x   = (const float*)d_in[0];
    const int*   eio = (const int*)  d_in[1];
    const int*   ei1 = (const int*)  d_in[2];
    const float* ea1 = (const float*)d_in[3];
    const int*   ei2 = (const int*)  d_in[4];
    const float* ea2 = (const float*)d_in[5];
    const float* W1  = (const float*)d_in[6];
    const float* b1  = (const float*)d_in[7];

    const int N    = in_sizes[0] / 256;     // 50000
    const int E_O  = in_sizes[1] / 2;       // 500000
    const int E1   = in_sizes[3];
    const int E2   = in_sizes[5];
    const int Etot = E_O + N;               // edges + self loops
    const int Mtiles = N / 16;              // 3125

    // ---- workspace carve-out (256B aligned slabs) ----
    char* ws = (char*)d_ws;
    size_t off = 0;
    auto carve = [&](size_t bytes) -> void* {
        void* p = ws + off; off += (bytes + 255) & ~(size_t)255; return p;
    };
    float*  h    = (float*) carve((size_t)N * 256 * 4);
    float*  ysum = (float*) carve((size_t)N * 16  * 4);
    float*  p    = (float*) carve((size_t)N * 256 * 4);
    float*  hg   = (float*) carve((size_t)N * 32  * 4);
    float*  asrc = (float*) carve((size_t)N * 2   * 4);
    float*  adst = (float*) carve((size_t)N * 2   * 4);
    float*  mx   = (float*) carve((size_t)N * 2   * 4);
    float*  den  = (float*) carve((size_t)N * 2   * 4);
    float*  ebuf = (float*) carve((size_t)Etot * 2 * 4);
    float*  gout = (float*) carve((size_t)N * 32  * 4);
    float*  ybuf = (float*) carve((size_t)N * 16  * 4);
    __bf16* xbf  = (__bf16*)carve((size_t)N * 256 * 2);   // bf16 A operands
    __bf16* pbf  = (__bf16*)carve((size_t)N * 256 * 2);
    __bf16* W1t  = (__bf16*)carve((size_t)256 * 256 * 2); // transposed bf16 B
    __bf16* gwt  = (__bf16*)carve((size_t)256 * 32  * 2);
    (void)ws_size;

    // ---- operand prep + linear1: h = x @ W1 + b1  (WMMA bf16) ----
    {
        long long n2 = (long long)N * 256 / 2;
        cvt_f32_bf16<<<(int)((n2 + 255) / 256), 256, 0, stream>>>(x, xbf, n2);
        cvt_f32_bf16_T<<<(256 * 256 + 255) / 256, 256, 0, stream>>>(W1, W1t, 256, 256);
        int waves  = Mtiles * 4;                       // Ngroups = 256/16/4
        int blocks = (waves + 7) / 8;
        wmma_gemm_bf16<4><<<blocks, 256, 0, stream>>>(xbf, W1t, b1, h,
                                                      Mtiles, 4, 256, 256, 256);
    }
    hipMemsetAsync(ysum, 0, (size_t)N * 16 * 4, stream);

    for (int lvl = 0; lvl < 2; ++lvl) {
        const int*   ei = lvl ? ei2 : ei1;
        const float* ea = lvl ? ea2 : ea1;
        const int    E  = lvl ? E2  : E1;
        const int    b  = 8 + lvl * 6;
        const float* gw = (const float*)d_in[b + 0];
        const float* as = (const float*)d_in[b + 1];
        const float* ad = (const float*)d_in[b + 2];
        const float* gb = (const float*)d_in[b + 3];
        const float* wm = (const float*)d_in[b + 4];
        const float* bm = (const float*)d_in[b + 5];

        // p = propagate(h)
        hipMemsetAsync(p, 0, (size_t)N * 256 * 4, stream);
        {
            long long tot = (long long)E * 64;
            propagate256<<<(int)((tot + 255) / 256), 256, 0, stream>>>(h, ei, ei + E, ea, p, E);
        }
        // hg = p @ Wg  (WMMA bf16, Ncols=32 -> NS=2, Ngroups=1)
        {
            long long n2 = (long long)N * 256 / 2;
            cvt_f32_bf16<<<(int)((n2 + 255) / 256), 256, 0, stream>>>(p, pbf, n2);
            cvt_f32_bf16_T<<<(256 * 32 + 255) / 256, 256, 0, stream>>>(gw, gwt, 256, 32);
            int waves  = Mtiles;
            int blocks = (waves + 7) / 8;
            wmma_gemm_bf16<2><<<blocks, 256, 0, stream>>>(pbf, gwt, nullptr, hg,
                                                          Mtiles, 1, 256, 256, 32);
        }
        gat_scores<<<(N * 2 + 255) / 256, 256, 0, stream>>>(hg, as, ad, asrc, adst, N);

        hipMemsetAsync(mx, 0xFF, (size_t)N * 2 * 4, stream);   // -> NaN, replaced on 1st max
        gat_edge_max<<<(Etot * 2 + 255) / 256, 256, 0, stream>>>(eio, E_O, Etot, asrc, adst, mx);

        hipMemsetAsync(den, 0, (size_t)N * 2 * 4, stream);
        gat_edge_exp<<<(Etot * 2 + 255) / 256, 256, 0, stream>>>(eio, E_O, Etot, asrc, adst,
                                                                 mx, ebuf, den);
        hipMemsetAsync(gout, 0, (size_t)N * 32 * 4, stream);
        {
            long long tot = (long long)Etot * 32;
            gat_aggregate<<<(int)((tot + 255) / 256), 256, 0, stream>>>(eio, E_O, Etot,
                                                                        hg, ebuf, den, gout);
        }
        // y = ELU(g + gb) @ Wm + bm
        mlp_elu<<<(N * 16 + 255) / 256, 256, 0, stream>>>(gout, gb, wm, bm, ybuf, N);
        // ysum += propagate(y)
        {
            long long tot = (long long)E * 16;
            propagate16<<<(int)((tot + 255) / 256), 256, 0, stream>>>(ybuf, ei, ei + E, ea, ysum, E);
        }
    }

    logsoftmax16<<<(N + 255) / 256, 256, 0, stream>>>(ysum, (float*)d_out, N);
}
